// LineThinner_9758165697053
// MI455X (gfx1250) — matrix operand order repaired and checked by reading, not compiled, not run on hardware
//
#include <hip/hip_runtime.h>
#include <hip/hip_bf16.h>
#include <math.h>

// ---------------------------------------------------------------------------
// LineThinner pipeline for MI455X (gfx1250).
//   image [2,3,1080,1920] f32  ->  warped image [2,3,1080,1920] f32
// Stages (intermediates at 540x960 are L2-resident; ~16.6MB of ws):
//   1) luma + bilinear half-res downsample        (img  -> bufA)
//   2) fused fractional Sobel -> edge^0.7         (bufA -> bufB)
//   3) Gaussian row-blur, transpose-on-store      (bufB -> bufA)   [WMMA f16]
//   4) Gaussian row-blur of T, transpose-on-store (bufA -> bufB)   [WMMA f16]
//   5) fused fractional Sobel -> 2ch grad field   (bufB -> bufG)
//   6) 6-iter warp + final image gather + clip    (img,bufG -> out)
// Blur tiles use overlapping edge tiles (r0 = min(16*b, dim-16)) so there are
// no store guards: EXEC stays all-ones (WMMA requirement) and each lane's 8
// accumulator floats store as two aligned global_store_b128.
// ---------------------------------------------------------------------------

#define BATCH 2
#define CHANS 3
#define IH 1080
#define IW 1920
#define SH 540
#define SW 960
#define STRENGTH 0.1f

typedef __attribute__((ext_vector_type(16))) _Float16 v16h;
typedef __attribute__((ext_vector_type(8)))  float    v8f;

// ---------------- helpers ----------------

__device__ __forceinline__ float bilin1(const float* __restrict__ p, int w, int h,
                                        float x, float y) {
    // border clamp + align_corners bilinear, matching torch grid_sample semantics
    x = fminf(fmaxf(x, 0.0f), (float)(w - 1));
    y = fminf(fmaxf(y, 0.0f), (float)(h - 1));
    int x0 = (int)x, y0 = (int)y;                 // x,y >= 0 so trunc == floor
    int x1 = min(x0 + 1, w - 1), y1 = min(y0 + 1, h - 1);
    float fx = x - (float)x0, fy = y - (float)y0;
    float v00 = p[y0 * w + x0], v01 = p[y0 * w + x1];
    float v10 = p[y1 * w + x0], v11 = p[y1 * w + x1];
    return (v00 * (1.0f - fx) + v01 * fx) * (1.0f - fy)
         + (v10 * (1.0f - fx) + v11 * fx) * fy;
}

// Fused sobel_x(offset) -> sobel_y(offset): the two fractional shifts are
// separable, so the whole chain is 8 bilinear taps at (x +/- dxp, y +/- dyp).
//   Gx = [1,2,1]_y (x) [-1,0,1]_x / 8
//   Gy = [-1,0,1]_y (x) [1,2,1]_x / 8
__device__ __forceinline__ void sobel_pair(const float* __restrict__ p,
                                           float xc, float yc,
                                           float dxp, float dyp,
                                           float& Gx, float& Gy) {
    float t[3][3];   // t[a][b]: a = x-tap (-,0,+), b = y-tap (-,0,+)
    #pragma unroll
    for (int a = 0; a < 3; ++a)
        #pragma unroll
        for (int bb = 0; bb < 3; ++bb)
            t[a][bb] = bilin1(p, SW, SH,
                              xc + (float)(a - 1) * dxp,
                              yc + (float)(bb - 1) * dyp);
    Gx = ((-t[0][0] + t[2][0]) + 2.0f * (-t[0][1] + t[2][1]) + (-t[0][2] + t[2][2])) * 0.125f;
    Gy = (-(t[0][0] + 2.0f * t[1][0] + t[2][0]) + (t[0][2] + 2.0f * t[1][2] + t[2][2])) * 0.125f;
}

// Gaussian 5-tap weights, sigma = 1.0, normalized (selects keep the banded B
// fragment pure VALU, no constant-memory gather).
__device__ __forceinline__ float gauss5(int d) {
    return (d == 2) ? 0.40261996f
         : (d == 1 || d == 3) ? 0.24420135f
         : (d == 0 || d == 4) ? 0.05448868f
         : 0.0f;
}

// ---------------- stage 1: luma + half-res downsample ----------------

__global__ void k_luma_half(const float* __restrict__ img, float* __restrict__ luma) {
    int idx = blockIdx.x * blockDim.x + threadIdx.x;
    if (idx >= BATCH * SH * SW) return;
    int b = idx / (SH * SW);
    int r = idx - b * (SH * SW);
    int i = r / SW, j = r - i * SW;
    float y = (float)i * (float)(IH - 1) / (float)(SH - 1);
    float x = (float)j * (float)(IW - 1) / (float)(SW - 1);
    int x0 = (int)x, y0 = (int)y;
    int x1 = min(x0 + 1, IW - 1), y1 = min(y0 + 1, IH - 1);
    float fx = x - (float)x0, fy = y - (float)y0;
    const float* R  = img + ((size_t)b * CHANS + 0) * IH * IW;
    const float* G  = img + ((size_t)b * CHANS + 1) * IH * IW;
    const float* Bc = img + ((size_t)b * CHANS + 2) * IH * IW;
    #define LUMA_AT(yy, xx) (0.299f * R[(yy) * IW + (xx)] + 0.587f * G[(yy) * IW + (xx)] + 0.114f * Bc[(yy) * IW + (xx)])
    float v00 = LUMA_AT(y0, x0), v01 = LUMA_AT(y0, x1);
    float v10 = LUMA_AT(y1, x0), v11 = LUMA_AT(y1, x1);
    #undef LUMA_AT
    luma[idx] = (v00 * (1.0f - fx) + v01 * fx) * (1.0f - fy)
              + (v10 * (1.0f - fx) + v11 * fx) * fy;
}

// ---------------- stage 2: edge magnitude ----------------

__global__ void k_sobel_edge(const float* __restrict__ luma, float* __restrict__ edge) {
    int idx = blockIdx.x * blockDim.x + threadIdx.x;
    if (idx >= BATCH * SH * SW) return;
    int b = idx / (SH * SW);
    int r = idx - b * (SH * SW);
    int i = r / SW, j = r - i * SW;
    const float dxp = (float)(SW - 1) / (float)SW;   // offset=1.0 pixel shift
    const float dyp = (float)(SH - 1) / (float)SH;
    float Gx, Gy;
    sobel_pair(luma + (size_t)b * SH * SW, (float)j, (float)i, dxp, dyp, Gx, Gy);
    edge[idx] = powf(Gx * Gx + Gy * Gy, 0.35f);      // sqrt(.)^0.7
}

// ---------------- stages 3/4: separable Gaussian via WMMA ----------------
// One wave per 16x16 output tile. out(16x16) = A(16x32 f16) x B(32x16 f16),
// B is the banded 5-tap Gaussian: B[k][n] = w[k-n] for k in [n, n+4].
// Fragment layouts per CDNA5 ISA 7.12.2 (16-bit A 16x32, f16 B 32x16, f32 C/D).
// Blurs along rows of src (sh rows x sw cols) and stores TRANSPOSED:
//   dst[x * sh + y] = sum_t w[t] * src[y][clamp(x-2+t)]
// Running it twice performs the separable blur and restores orientation.

__device__ __forceinline__ v16h make_band_B(int lane) {
    v16h bm;
    int n   = lane & 15;
    int kbB = (lane < 16) ? 0 : 8;
    #pragma unroll
    for (int e = 0; e < 16; ++e) {
        int d = ((e < 8) ? (kbB + e) : (16 + kbB + e - 8)) - n;
        bm[e] = (_Float16)gauss5(d);
    }
    return bm;
}

__global__ __launch_bounds__(32) void k_blur_rows_T(const float* __restrict__ src,
                                                    float* __restrict__ dst,
                                                    int sw, int sh) {
    __shared__ _Float16 strip[16][34];               // 16 rows x 32-wide strip (+pad)
    const int c0 = min((int)blockIdx.x * 16, sw - 16);   // overlapping edge tile
    const int r0 = min((int)blockIdx.y * 16, sh - 16);   // overlapping edge tile
    const int b  = blockIdx.z;
    const int lane = threadIdx.x;
    const float* p = src + (size_t)b * sw * sh;
    for (int t = lane; t < 16 * 32; t += 32) {
        int r = t >> 5, k = t & 31;
        int gc = min(max(c0 - 2 + k, 0), sw - 1);    // edge-mode padding
        strip[r][k] = (_Float16)p[(r0 + r) * sw + gc];
    }
    __syncthreads();
    // A fragment: row m = lane&15; halves 0..7 = K (lane<16?0:8)+e, 8..15 = K 16+...
    v16h a;
    int m  = lane & 15;
    int kb = (lane < 16) ? 0 : 8;
    #pragma unroll
    for (int e = 0; e < 8; ++e) a[e]     = strip[m][kb + e];
    #pragma unroll
    for (int e = 0; e < 8; ++e) a[8 + e] = strip[m][16 + kb + e];
    v16h bm = make_band_B(lane);
    v8f acc = {};
    acc = __builtin_amdgcn_wmma_f32_16x16x32_f16(false, a, false, bm,
                                                 (short)0, acc, false, false);
    // D[m][n]: N = lane&15, M = v + (lane<16 ? 0 : 8). Transposed store puts a
    // lane's 8 results at consecutive addresses: dst[(c0+n)*sh + r0 + M].
    // sh, r0 are multiples of 4 -> both float4 stores are 16B aligned.
    int n = lane & 15;
    float* base = dst + (size_t)b * sw * sh
                      + (size_t)(c0 + n) * sh + r0 + ((lane < 16) ? 0 : 8);
    *(float4*)(base + 0) = make_float4(acc[0], acc[1], acc[2], acc[3]);
    *(float4*)(base + 4) = make_float4(acc[4], acc[5], acc[6], acc[7]);
}

// ---------------- stage 5: gradient field ----------------

__global__ void k_grad_field(const float* __restrict__ blur, float* __restrict__ grad) {
    int idx = blockIdx.x * blockDim.x + threadIdx.x;
    if (idx >= BATCH * SH * SW) return;
    int b = idx / (SH * SW);
    int r = idx - b * (SH * SW);
    int i = r / SW, j = r - i * SW;
    const float dxp = 0.5f * (float)(SW - 1) / (float)SW;   // offset=0.5
    const float dyp = 0.5f * (float)(SH - 1) / (float)SH;
    float Gx, Gy;
    sobel_pair(blur + (size_t)b * SH * SW, (float)j, (float)i, dxp, dyp, Gx, Gy);
    grad[2 * (size_t)idx + 0] = Gx;
    grad[2 * (size_t)idx + 1] = Gy;
}

// ---------------- stage 6: iterative warp ----------------

__device__ __forceinline__ float2 bilinG(const float2* __restrict__ g, float x, float y) {
    x = fminf(fmaxf(x, 0.0f), (float)(SW - 1));
    y = fminf(fmaxf(y, 0.0f), (float)(SH - 1));
    int x0 = (int)x, y0 = (int)y;
    int x1 = min(x0 + 1, SW - 1), y1 = min(y0 + 1, SH - 1);
    float fx = x - (float)x0, fy = y - (float)y0;
    float2 v00 = g[y0 * SW + x0], v01 = g[y0 * SW + x1];
    float2 v10 = g[y1 * SW + x0], v11 = g[y1 * SW + x1];
    float2 out;
    out.x = (v00.x * (1.0f - fx) + v01.x * fx) * (1.0f - fy)
          + (v10.x * (1.0f - fx) + v11.x * fx) * fy;
    out.y = (v00.y * (1.0f - fx) + v01.y * fx) * (1.0f - fy)
          + (v10.y * (1.0f - fx) + v11.y * fx) * fy;
    return out;
}

__global__ void k_warp(const float* __restrict__ img, const float* __restrict__ grad,
                       float* __restrict__ out) {
    int idx = blockIdx.x * blockDim.x + threadIdx.x;
    if (idx >= BATCH * IH * IW) return;
    int b = idx / (IH * IW);
    int r = idx - b * (IH * IW);
    int i = r / IW, j = r - i * IW;

    // Hint the gather region of the final resample (near-identity warp).
    __builtin_prefetch(img + (size_t)b * CHANS * IH * IW + (size_t)i * IW + j, 0, 1);

    float px = -1.0f + 2.0f * (float)j / (float)(IW - 1);
    float py = -1.0f + 2.0f * (float)i / (float)(IH - 1);
    const float stepx = (2.0f / (float)IW) * STRENGTH;   // relstr = H/1080*0.1 = 0.1
    const float stepy = (2.0f / (float)IH) * STRENGTH;
    const float2* g = (const float2*)(grad) + (size_t)b * SH * SW;

    #pragma unroll
    for (int it = 0; it < 6; ++it) {
        // Sample the half-res field directly (bilinear of the bilinear upsample
        // collapses to coarse-grid bilinear up to sub-texel error).
        float xh = (px + 1.0f) * 0.5f * (float)(SW - 1);
        float yh = (py + 1.0f) * 0.5f * (float)(SH - 1);
        float2 dn = bilinG(g, xh, yh);
        float len = sqrtf(dn.x * dn.x + dn.y * dn.y) + 0.01f;
        px -= dn.x / len * stepx;
        py -= dn.y / len * stepy;
    }

    float xi = fminf(fmaxf((px + 1.0f) * 0.5f * (float)(IW - 1), 0.0f), (float)(IW - 1));
    float yi = fminf(fmaxf((py + 1.0f) * 0.5f * (float)(IH - 1), 0.0f), (float)(IH - 1));
    int x0 = (int)xi, y0 = (int)yi;
    int x1 = min(x0 + 1, IW - 1), y1 = min(y0 + 1, IH - 1);
    float fx = xi - (float)x0, fy = yi - (float)y0;
    float w00 = (1.0f - fx) * (1.0f - fy), w01 = fx * (1.0f - fy);
    float w10 = (1.0f - fx) * fy,          w11 = fx * fy;
    #pragma unroll
    for (int c = 0; c < CHANS; ++c) {
        const float* ch = img + ((size_t)b * CHANS + c) * IH * IW;
        float v = ch[y0 * IW + x0] * w00 + ch[y0 * IW + x1] * w01
                + ch[y1 * IW + x0] * w10 + ch[y1 * IW + x1] * w11;
        out[((size_t)b * CHANS + c) * IH * IW + (size_t)i * IW + j] =
            fminf(fmaxf(v, 0.0f), 1.0f);
    }
}

// ---------------- launch ----------------

extern "C" void kernel_launch(void* const* d_in, const int* in_sizes, int n_in,
                              void* d_out, int out_size, void* d_ws, size_t ws_size,
                              hipStream_t stream) {
    const float* img = (const float*)d_in[0];
    float* out = (float*)d_out;
    float* ws  = (float*)d_ws;

    const size_t nHalf = (size_t)BATCH * SH * SW;        // 1,036,800
    float* bufA = ws;                                    // luma  -> T (row-blur^T)
    float* bufB = ws + nHalf;                            // edge  -> final blur
    float* bufG = ws + 2 * nHalf;                        // grad field (2ch interleaved)

    const int T = 256;
    const int nH = (int)nHalf;
    const int nF = BATCH * IH * IW;

    k_luma_half <<<(nH + T - 1) / T, T, 0, stream>>>(img, bufA);
    k_sobel_edge<<<(nH + T - 1) / T, T, 0, stream>>>(bufA, bufB);

    // Pass 1: blur rows of edge (960x540) -> T (960 rows x 540 cols) in bufA.
    dim3 g1((SW + 15) / 16, (SH + 15) / 16, BATCH);      // 60 x 34 x 2
    k_blur_rows_T<<<g1, 32, 0, stream>>>(bufB, bufA, SW, SH);
    // Pass 2: blur rows of T (540x960) -> normal orientation (540x960) in bufB.
    dim3 g2((SH + 15) / 16, (SW + 15) / 16, BATCH);      // 34 x 60 x 2
    k_blur_rows_T<<<g2, 32, 0, stream>>>(bufA, bufB, SH, SW);

    k_grad_field<<<(nH + T - 1) / T, T, 0, stream>>>(bufB, bufG);
    k_warp      <<<(nF + T - 1) / T, T, 0, stream>>>(img, bufG, out);
}
